// CrossAttentionHead_76467597738151
// MI455X (gfx1250) — compile-verified, matrix operand order checked by reading, not run on hardware
//
#include <hip/hip_runtime.h>
#include <hip/hip_bf16.h>
#include <math.h>

typedef __attribute__((ext_vector_type(16))) __bf16 v16bf;
typedef __attribute__((ext_vector_type(8)))  float  v8f;

#define EMBED 768
#define HEAD  128
#define SEQ   2048
#define BATCH 8
#define ROWS  (BATCH * SEQ)          // 16384
#define ROWTILES (ROWS / 16)         // 1024 (16-row tiles)

// Compiler-level ordering fence: same-wave LDS ops execute in order in HW
// (CDNA5 ISA: LDS tracked by DScnt, in-order); this just stops the compiler
// from reordering the store/load sequences around it.
#define WAVE_LDS_FENCE() asm volatile("" ::: "memory")

// ---------------------------------------------------------------------------
// W f32[EMBED][HEAD] -> Wt bf16[HEAD][EMBED], scaled (folds softmax scale in)
// ---------------------------------------------------------------------------
__global__ __launch_bounds__(256) void wconv_kernel(
    const float* __restrict__ W, __bf16* __restrict__ Wt, float scale)
{
  int idx = blockIdx.x * 256 + threadIdx.x;        // 0 .. EMBED*HEAD-1
  int k = idx / HEAD;
  int n = idx % HEAD;
  Wt[(size_t)n * EMBED + k] = (__bf16)(W[idx] * scale);
}

// ---------------------------------------------------------------------------
// Projection: Y = X @ Wt' + bias*bscale, bf16 out (scale pre-folded into Wt).
// One wave computes a 16x128 tile; 24 K-steps of wmma_f32_16x16x32_bf16 x 8.
// transpose_out = 0: Y[r][n] row-major  (Q, K)
// transpose_out = 1: Y[b][n][s] col-major per batch (V -> Vt for P@V B-frags)
// ---------------------------------------------------------------------------
__global__ __launch_bounds__(128) void proj_kernel(
    const float*  __restrict__ X,    // [ROWS][EMBED]
    const __bf16* __restrict__ Wt,   // [HEAD][EMBED] (transposed, scaled bf16)
    const float*  __restrict__ bias, // [HEAD]
    __bf16* __restrict__ Y,
    float bscale, int transpose_out)
{
  const int lane = threadIdx.x & 31;
  const int wave = threadIdx.x >> 5;
  const int m0   = (blockIdx.x * 4 + wave) * 16;
  const int r    = lane & 15;
  const int hi   = lane >> 4;
  const int kb   = hi * 8;           // K sub-base per ISA 16-bit A/B layout

  v8f acc[8];
  #pragma unroll
  for (int t = 0; t < 8; ++t) acc[t] = (v8f)(0.0f);

  for (int k0 = 0; k0 < EMBED; k0 += 32) {
    // A fragment: row m0+r; element e holds K = kb + (e<8 ? e : e+8)
    v16bf a;
    const float* xrow = X + (size_t)(m0 + r) * EMBED + k0;
    #pragma unroll
    for (int j = 0; j < 8; ++j) {
      a[j]     = (__bf16)xrow[kb + j];
      a[8 + j] = (__bf16)xrow[16 + kb + j];
    }
    // Preload all 8 B fragments (distinct regs -> one clause, staggered waits)
    v16bf bf_[8];
    #pragma unroll
    for (int t = 0; t < 8; ++t) {
      const __bf16* wrow = Wt + (size_t)(t * 16 + r) * EMBED + k0;
      #pragma unroll
      for (int j = 0; j < 8; ++j) {
        bf_[t][j]     = wrow[kb + j];
        bf_[t][8 + j] = wrow[16 + kb + j];
      }
    }
    #pragma unroll
    for (int t = 0; t < 8; ++t)
      acc[t] = __builtin_amdgcn_wmma_f32_16x16x32_bf16(
          false, a, false, bf_[t], (short)0, acc[t], false, false);
  }

  // C layout: VGPR rr, lane -> row (m0 + 8*hi + rr), col (t*16 + r)
  if (!transpose_out) {
    #pragma unroll
    for (int t = 0; t < 8; ++t) {
      float bn = bias[t * 16 + r] * bscale;
      #pragma unroll
      for (int rr = 0; rr < 8; ++rr)
        Y[(size_t)(m0 + 8 * hi + rr) * HEAD + t * 16 + r] =
            (__bf16)(acc[t][rr] + bn);
    }
  } else {
    const int b = m0 / SEQ;            // tiles never straddle a batch
    const int s0 = (m0 % SEQ) + 8 * hi;
    #pragma unroll
    for (int t = 0; t < 8; ++t) {
      float bn = bias[t * 16 + r] * bscale;
      __bf16* ycol = Y + ((size_t)b * HEAD + t * 16 + r) * SEQ + s0;
      #pragma unroll
      for (int rr = 0; rr < 8; ++rr)
        ycol[rr] = (__bf16)(acc[t][rr] + bn);
    }
  }
}

// ---------------------------------------------------------------------------
// Flash attention: one wave per 16-row query tile, KV chunks of 32,
// online softmax, all matmuls via v_wmma_f32_16x16x32_bf16.
// Waves are fully independent (private LDS slices, no block barriers).
// ---------------------------------------------------------------------------
__global__ __launch_bounds__(128) void attn_kernel(
    const __bf16* __restrict__ Qb,  // [BATCH][SEQ][HEAD] (scale folded in)
    const __bf16* __restrict__ Kb,  // [BATCH][SEQ][HEAD]
    const __bf16* __restrict__ Vt,  // [BATCH][HEAD][SEQ] (transposed)
    float* __restrict__ Out)        // [BATCH][SEQ][HEAD] f32
{
  __shared__ float  lds_S[4][16][33];   // S tile, padded rows
  __shared__ __bf16 lds_P[4][16][32];   // P tile (bf16 probabilities)
  __shared__ float  lds_f[4][16];       // per-row rescale factor
  __shared__ float  lds_l[4][16];       // per-row 1/l at the end

  const int lane = threadIdx.x & 31;
  const int wave = threadIdx.x >> 5;
  const int qt   = blockIdx.x * 4 + wave;       // 0..1023
  const int b    = qt >> 7;                     // 128 tiles per batch
  const int q0   = (qt & 127) * 16;

  const __bf16* Qp  = Qb + (size_t)b * SEQ * HEAD;
  const __bf16* Kp  = Kb + (size_t)b * SEQ * HEAD;
  const __bf16* Vtp = Vt + (size_t)b * HEAD * SEQ;

  const int r  = lane & 15;
  const int hi = lane >> 4;
  const int kb = hi * 8;

  // Q tile as 4 A-fragments (head-dim chunks of 32), kept resident
  v16bf qf[4];
  {
    const __bf16* qrow = Qp + (size_t)(q0 + r) * HEAD;
    #pragma unroll
    for (int c = 0; c < 4; ++c) {
      #pragma unroll
      for (int j = 0; j < 8; ++j) {
        qf[c][j]     = qrow[c * 32 + kb + j];
        qf[c][8 + j] = qrow[c * 32 + 16 + kb + j];
      }
    }
  }

  v8f o[8];
  #pragma unroll
  for (int t = 0; t < 8; ++t) o[t] = (v8f)(0.0f);
  float m_i = -3.0e38f;   // running row max (valid in lanes 0..15, row = lane)
  float l_i = 0.0f;       // running row sum

  for (int kv0 = 0; kv0 < SEQ; kv0 += 32) {
    // prefetch next chunk's K rows and Vt segments into cache
    if (kv0 + 32 < SEQ) {
      __builtin_prefetch(Kp + (size_t)(kv0 + 32 + lane) * HEAD, 0, 3);
      __builtin_prefetch(Vtp + (size_t)(lane * 4) * SEQ + kv0 + 32, 0, 3);
    }

    // ---- S = Q @ K^T for this 16x32 chunk (scale folded into Q) ----
    v8f s0 = (v8f)(0.0f), s1 = (v8f)(0.0f);
    #pragma unroll
    for (int h = 0; h < 2; ++h) {
      const __bf16* krow = Kp + (size_t)(kv0 + h * 16 + r) * HEAD;
      v16bf kf[4];                       // preload: one clause, staggered waits
      #pragma unroll
      for (int c = 0; c < 4; ++c) {
        #pragma unroll
        for (int j = 0; j < 8; ++j) {
          kf[c][j]     = krow[c * 32 + kb + j];
          kf[c][8 + j] = krow[c * 32 + 16 + kb + j];
        }
      }
      #pragma unroll
      for (int c = 0; c < 4; ++c) {
        if (h == 0)
          s0 = __builtin_amdgcn_wmma_f32_16x16x32_bf16(
              false, qf[c], false, kf[c], (short)0, s0, false, false);
        else
          s1 = __builtin_amdgcn_wmma_f32_16x16x32_bf16(
              false, qf[c], false, kf[c], (short)0, s1, false, false);
      }
    }
    // spill S to LDS (C layout -> plain [16][32]); same-wave LDS is in-order
    #pragma unroll
    for (int rr = 0; rr < 8; ++rr) {
      lds_S[wave][8 * hi + rr][r]      = s0[rr];
      lds_S[wave][8 * hi + rr][16 + r] = s1[rr];
    }
    WAVE_LDS_FENCE();

    // ---- online softmax: lane i (<16) owns query row i ----
    if (lane < 16) {
      float sv[32];
      float mc = -3.0e38f;
      #pragma unroll
      for (int j = 0; j < 32; ++j) {
        sv[j] = lds_S[wave][lane][j];
        mc = fmaxf(mc, sv[j]);
      }
      float mn = fmaxf(m_i, mc);
      float f  = __expf(m_i - mn);
      float sum = 0.0f;
      #pragma unroll
      for (int j = 0; j < 32; ++j) {
        float p = __expf(sv[j] - mn);
        sum += p;
        lds_P[wave][lane][j] = (__bf16)p;
      }
      m_i = mn;
      l_i = l_i * f + sum;
      lds_f[wave][lane] = f;
    }
    WAVE_LDS_FENCE();

    // ---- rescale O accumulators by per-row factor ----
    #pragma unroll
    for (int rr = 0; rr < 8; ++rr) {
      float f = lds_f[wave][8 * hi + rr];
      #pragma unroll
      for (int t = 0; t < 8; ++t) o[t][rr] *= f;
    }

    // ---- P (16x32) as A-fragment from LDS ----
    v16bf pf;
    #pragma unroll
    for (int j = 0; j < 8; ++j) {
      pf[j]     = lds_P[wave][r][kb + j];
      pf[8 + j] = lds_P[wave][r][16 + kb + j];
    }

    // ---- O += P @ V : B-fragments contiguous from transposed Vt ----
    #pragma unroll
    for (int g = 0; g < 2; ++g) {
      v16bf vf[4];                       // preload 4 tiles per group
      #pragma unroll
      for (int t = 0; t < 4; ++t) {
        const __bf16* vrow = Vtp + (size_t)((g * 4 + t) * 16 + r) * SEQ + kv0;
        #pragma unroll
        for (int j = 0; j < 8; ++j) {
          vf[t][j]     = vrow[kb + j];
          vf[t][8 + j] = vrow[16 + kb + j];
        }
      }
      #pragma unroll
      for (int t = 0; t < 4; ++t)
        o[g * 4 + t] = __builtin_amdgcn_wmma_f32_16x16x32_bf16(
            false, pf, false, vf[t], (short)0, o[g * 4 + t], false, false);
    }
    WAVE_LDS_FENCE();
  }

  // ---- finalize: divide rows by l and store f32 ----
  if (lane < 16) lds_l[wave][lane] = 1.0f / l_i;
  WAVE_LDS_FENCE();

  float* orow = Out + ((size_t)b * SEQ + q0 + 8 * hi) * HEAD;
  #pragma unroll
  for (int rr = 0; rr < 8; ++rr) {
    float linv = lds_l[wave][8 * hi + rr];
    #pragma unroll
    for (int t = 0; t < 8; ++t)
      orow[(size_t)rr * HEAD + t * 16 + r] = o[t][rr] * linv;
  }
}

// ---------------------------------------------------------------------------
extern "C" void kernel_launch(void* const* d_in, const int* in_sizes, int n_in,
                              void* d_out, int out_size, void* d_ws, size_t ws_size,
                              hipStream_t stream) {
  const float* x  = (const float*)d_in[0];
  // d_in[1] = enc_output: unused by the faithful reference
  const float* Wq = (const float*)d_in[2];
  const float* bq = (const float*)d_in[3];
  const float* Wk = (const float*)d_in[4];
  const float* bk = (const float*)d_in[5];
  const float* Wv = (const float*)d_in[6];
  const float* bv = (const float*)d_in[7];
  float* out = (float*)d_out;

  const size_t qkv_elems = (size_t)ROWS * HEAD;       // 2,097,152
  const size_t w_elems   = (size_t)EMBED * HEAD;      // 98,304
  __bf16* Qb  = (__bf16*)d_ws;
  __bf16* Kb  = Qb + qkv_elems;
  __bf16* Vt  = Kb + qkv_elems;
  __bf16* Wtq = Vt + qkv_elems;
  __bf16* Wtk = Wtq + w_elems;
  __bf16* Wtv = Wtk + w_elems;

  const float scale = 1.0f / sqrtf((float)EMBED);

  dim3 blk(128);
  dim3 grd(ROWTILES / 4);                 // 256 blocks, 4 waves each
  dim3 wblk(256), wgrd(w_elems / 256);    // 384 blocks

  wconv_kernel<<<wgrd, wblk, 0, stream>>>(Wq, Wtq, scale);  // fold softmax scale
  wconv_kernel<<<wgrd, wblk, 0, stream>>>(Wk, Wtk, 1.0f);
  wconv_kernel<<<wgrd, wblk, 0, stream>>>(Wv, Wtv, 1.0f);

  proj_kernel<<<grd, blk, 0, stream>>>(x, Wtq, bq, Qb, scale, 0);
  proj_kernel<<<grd, blk, 0, stream>>>(x, Wtk, bk, Kb, 1.0f, 0);
  proj_kernel<<<grd, blk, 0, stream>>>(x, Wtv, bv, Vt, 1.0f, 1);

  attn_kernel<<<grd, blk, 0, stream>>>(Qb, Kb, Vt, out);
}
